// MultiHeadAttention_16260746182816
// MI455X (gfx1250) — compile-verified, matrix operand order checked by reading, not run on hardware
//
#include <hip/hip_runtime.h>
#include <hip/hip_bf16.h>
#include <math.h>

// ---------------- types ----------------
typedef __attribute__((ext_vector_type(16))) __bf16 v16bf;
typedef __attribute__((ext_vector_type(8)))  __bf16 v8bf;
typedef __attribute__((ext_vector_type(4)))  __bf16 v4bf;
typedef __attribute__((ext_vector_type(8)))  float  v8f;
typedef __attribute__((ext_vector_type(4)))  float  v4f;

#define B_  4
#define T_  2048
#define C_  1024
#define H_  16
#define DH_ 64
#define C3_ 3072

// ---------------- WMMA helpers ----------------
static __device__ __forceinline__ v8f wmma_bf16(v16bf a, v16bf b, v8f c) {
    // D(f32 16x16) = A(bf16 16x32) x B(bf16 32x16) + C
    return __builtin_amdgcn_wmma_f32_16x16x32_bf16(
        /*neg_a=*/false, a, /*neg_b=*/false, b,
        /*c_mod=*/(short)0, c, /*reuse_a=*/false, /*reuse_b=*/false);
}

// Load a 16x32 bf16 fragment (A layout; identical pattern serves as B when the
// source array is stored [N][K] row-major). Element (r,k) lives at base[r*ld+k].
// Per ISA 16-bit A layout: lane l -> r = l%16, kh = (l/16)*8;
// vector elems [0..7] = K kh..kh+7, elems [8..15] = K 16+kh..16+kh+7.
static __device__ __forceinline__ v16bf load_frag(const __bf16* base, int ld) {
    int lane = threadIdx.x & 31;
    int r    = lane & 15;
    int kh   = (lane >> 4) << 3;          // 0 or 8
    const __bf16* p = base + (size_t)r * ld;
    v8bf lo = *(const v8bf*)(p + kh);
    v8bf hi = *(const v8bf*)(p + 16 + kh);
    v16bf out;
#pragma unroll
    for (int i = 0; i < 8; ++i) { out[i] = lo[i]; out[i + 8] = hi[i]; }
    return out;
}

// ---------------- stage 0: conversions ----------------
__global__ void k_cast_bf16(const float* __restrict__ in, __bf16* __restrict__ out, int n) {
    int i = blockIdx.x * blockDim.x + threadIdx.x;
    if (i < n) out[i] = (__bf16)in[i];
}

// W[K][N] (fp32, [in,out]) -> Wt[N][K] (bf16)
__global__ void k_transpose_cast(const float* __restrict__ W, __bf16* __restrict__ Wt,
                                 int K, int N) {
    int i = blockIdx.x * blockDim.x + threadIdx.x;
    if (i < K * N) {
        int k = i % K;
        int n = i / K;
        Wt[(size_t)n * K + k] = (__bf16)W[(size_t)k * N + n];
    }
}

// V slice of qkv ([B*T][3C] bf16, cols [2C,3C)) -> Vt[(b*H+h)*DH + d][T]
__global__ void k_vt(const __bf16* __restrict__ qkv, __bf16* __restrict__ Vt) {
    int i = blockIdx.x * blockDim.x + threadIdx.x;   // total = B*H*DH*T = 2^23
    int t  = i & (T_ - 1);
    int d  = (i >> 11) & (DH_ - 1);
    int bh = i >> 17;                                 // b*H + h
    int b  = bh >> 4, h = bh & 15;
    Vt[i] = qkv[(size_t)(b * T_ + t) * C3_ + 2 * C_ + h * DH_ + d];
}

// ---------------- tiled WMMA GEMM: out = A[M,K] * Bt[N,K]^T + bias ----------------
// block = 128 threads (4 waves); wave tile = 32x64; block tile = 64x128.
template <bool OUT_BF16>
__global__ void k_gemm_bias(const __bf16* __restrict__ A, const __bf16* __restrict__ Bt,
                            const float* __restrict__ bias, void* __restrict__ outv,
                            int M, int N, int K) {
    int w    = threadIdx.x >> 5;
    int wm   = w & 1, wn = w >> 1;
    int bm   = blockIdx.y * 64 + wm * 32;
    int bn   = blockIdx.x * 128 + wn * 64;
    int lane = threadIdx.x & 31;
    int n16  = lane & 15;
    int hi8  = (lane >> 4) << 3;

    v8f acc[2][4];
#pragma unroll
    for (int i = 0; i < 2; ++i)
#pragma unroll
        for (int j = 0; j < 4; ++j) { v8f z = {}; acc[i][j] = z; }

    for (int kk = 0; kk < K; kk += 32) {
        v16bf a0 = load_frag(A + (size_t)bm * K + kk, K);
        v16bf a1 = load_frag(A + (size_t)(bm + 16) * K + kk, K);
#pragma unroll
        for (int j = 0; j < 4; ++j) {
            v16bf bb = load_frag(Bt + (size_t)(bn + j * 16) * K + kk, K);
            acc[0][j] = wmma_bf16(a0, bb, acc[0][j]);
            acc[1][j] = wmma_bf16(a1, bb, acc[1][j]);
        }
    }

#pragma unroll
    for (int i = 0; i < 2; ++i)
#pragma unroll
        for (int j = 0; j < 4; ++j) {
            int   col = bn + j * 16 + n16;
            float bv  = bias[col];
#pragma unroll
            for (int r = 0; r < 8; ++r) {
                int   row = bm + i * 16 + r + hi8;
                float v   = acc[i][j][r] + bv;
                if (OUT_BF16) ((__bf16*)outv)[(size_t)row * N + col] = (__bf16)v;
                else          ((float*)outv)[(size_t)row * N + col]  = v;
            }
        }
}

// ---------------- flash attention ----------------
// grid = B*H*(T/64); block = 128 threads (4 waves); wave owns 16 query rows.
__global__ void k_attention(const __bf16* __restrict__ qkv, const __bf16* __restrict__ Vt,
                            __bf16* __restrict__ outO) {
    __shared__ __align__(16) float  Ssh[4][16 * 128];
    __shared__ __align__(16) __bf16 Psh[4][16 * 128];

    int bid = blockIdx.x;
    int qt  = bid & 31;                 // T/64 = 32 query tiles
    int h   = (bid >> 5) & 15;
    int b   = bid >> 9;

    int w    = threadIdx.x >> 5;
    int lane = threadIdx.x & 31;
    int n16  = lane & 15;
    int hi8  = (lane >> 4) << 3;
    int half = lane >> 4;

    const float scale = 0.125f;         // 1/sqrt(64)

    const __bf16* Qb = qkv + (size_t)(b * T_ + qt * 64 + w * 16) * C3_ + h * DH_;
    const __bf16* Kb = qkv + (size_t)(b * T_) * C3_ + C_ + h * DH_;
    const __bf16* Vb = Vt + (size_t)((b * H_ + h) * DH_) * T_;

    v16bf qf0 = load_frag(Qb + 0, C3_);
    v16bf qf1 = load_frag(Qb + 32, C3_);

    v8f o[4];
#pragma unroll
    for (int d = 0; d < 4; ++d) { v8f z = {}; o[d] = z; }

    float m_run = -1.0e30f;
    float l_run = 0.0f;

    float*  Sw = &Ssh[w][0];
    __bf16* Pw = &Psh[w][0];

    for (int kc = 0; kc < T_ / 128; ++kc) {
        int kbase = kc * 128;

        // prefetch next chunk's K rows (global_prefetch_b8)
        if (kc + 1 < T_ / 128) {
            __builtin_prefetch((const void*)(Kb + (size_t)(kbase + 128 + lane * 4) * C3_), 0, 0);
            __builtin_prefetch((const void*)(Vb + (size_t)(lane * 2) * T_ + kbase + 128), 0, 0);
        }

        // ---- S = Q K^T (16 x 128), one 16x16 accumulator at a time ----
#pragma unroll
        for (int j = 0; j < 8; ++j) {
            v8f sj = {};
            v16bf b0 = load_frag(Kb + (size_t)(kbase + j * 16) * C3_ + 0, C3_);
            sj = wmma_bf16(qf0, b0, sj);
            v16bf b1 = load_frag(Kb + (size_t)(kbase + j * 16) * C3_ + 32, C3_);
            sj = wmma_bf16(qf1, b1, sj);
            // C layout: lane col = n16, rows = r + hi8
#pragma unroll
            for (int r = 0; r < 8; ++r)
                Sw[(r + hi8) * 128 + j * 16 + n16] = sj[r] * scale;
        }

        __syncthreads();

        // ---- online softmax: lane owns row n16, columns [half*64, half*64+64) ----
        int rbase = n16 * 128 + half * 64;                 // 256B-aligned
        const v4f* Sr4 = (const v4f*)(Sw + rbase);
        float mx = -1.0e30f;
#pragma unroll
        for (int i = 0; i < 16; ++i) {
            v4f v = Sr4[i];
            mx = fmaxf(mx, fmaxf(fmaxf(v[0], v[1]), fmaxf(v[2], v[3])));
        }
        mx = fmaxf(mx, __shfl_xor(mx, 16, 32));
        float mnew  = fmaxf(m_run, mx);
        float alpha = __expf(m_run - mnew);
        float lsum  = 0.0f;
        v4bf* Pr4 = (v4bf*)(Pw + rbase);
#pragma unroll
        for (int i = 0; i < 16; ++i) {
            v4f v = Sr4[i];
            float e0 = __expf(v[0] - mnew);
            float e1 = __expf(v[1] - mnew);
            float e2 = __expf(v[2] - mnew);
            float e3 = __expf(v[3] - mnew);
            v4bf pv;
            pv[0] = (__bf16)e0; pv[1] = (__bf16)e1; pv[2] = (__bf16)e2; pv[3] = (__bf16)e3;
            Pr4[i] = pv;
            lsum += (e0 + e1) + (e2 + e3);
        }
        lsum += __shfl_xor(lsum, 16, 32);
        l_run = l_run * alpha + lsum;
        m_run = mnew;

        // rescale O accumulators by per-row alpha
#pragma unroll
        for (int d = 0; d < 4; ++d)
#pragma unroll
            for (int r = 0; r < 8; ++r)
                o[d][r] *= __shfl(alpha, r + hi8, 32);

        __syncthreads();

        // ---- O += P V  (P: 16x128 bf16 in LDS, V^T: [DH][T]) ----
#pragma unroll
        for (int ks = 0; ks < 4; ++ks) {
            v16bf pa = load_frag(Pw + ks * 32, 128);
#pragma unroll
            for (int d = 0; d < 4; ++d) {
                v16bf vb = load_frag(Vb + (size_t)(d * 16) * T_ + kbase + ks * 32, T_);
                o[d] = wmma_bf16(pa, vb, o[d]);
            }
        }
    }

    // ---- normalize and store bf16 [B*T][C] ----
    float inv = 1.0f / l_run;
#pragma unroll
    for (int d = 0; d < 4; ++d)
#pragma unroll
        for (int r = 0; r < 8; ++r) {
            float v   = o[d][r] * __shfl(inv, r + hi8, 32);
            int   row = b * T_ + qt * 64 + w * 16 + r + hi8;
            int   col = h * DH_ + d * 16 + n16;
            outO[(size_t)row * C_ + col] = (__bf16)v;
        }
}

// ---------------- launch ----------------
extern "C" void kernel_launch(void* const* d_in, const int* in_sizes, int n_in,
                              void* d_out, int out_size, void* d_ws, size_t ws_size,
                              hipStream_t stream) {
    const float* X      = (const float*)d_in[0];
    const float* W_qkv  = (const float*)d_in[1];
    const float* b_qkv  = (const float*)d_in[2];
    const float* W_proj = (const float*)d_in[3];
    const float* b_proj = (const float*)d_in[4];
    float* out = (float*)d_out;

    char* ws = (char*)d_ws;
    size_t off = 0;
    __bf16* Xbf  = (__bf16*)(ws + off); off += (size_t)B_ * T_ * C_ * 2;        // 16 MiB
    __bf16* Wqt  = (__bf16*)(ws + off); off += (size_t)C3_ * C_ * 2;            // 6 MiB
    __bf16* Wpt  = (__bf16*)(ws + off); off += (size_t)C_ * C_ * 2;             // 2 MiB
    __bf16* qkv  = (__bf16*)(ws + off); off += (size_t)B_ * T_ * C3_ * 2;       // 48 MiB
    __bf16* Vt   = (__bf16*)(ws + off); off += (size_t)B_ * H_ * DH_ * T_ * 2;  // 16 MiB
    __bf16* AttO = (__bf16*)(ws + off); off += (size_t)B_ * T_ * C_ * 2;        // 16 MiB

    const int M = B_ * T_;           // 8192

    // 0) conversions
    {
        int n = M * C_;
        k_cast_bf16<<<(n + 255) / 256, 256, 0, stream>>>(X, Xbf, n);
        int nw = C_ * C3_;
        k_transpose_cast<<<(nw + 255) / 256, 256, 0, stream>>>(W_qkv, Wqt, C_, C3_);
        int np = C_ * C_;
        k_transpose_cast<<<(np + 255) / 256, 256, 0, stream>>>(W_proj, Wpt, C_, C_);
    }

    // 1) qkv = X * W_qkv + b_qkv   (bf16 out)
    k_gemm_bias<true><<<dim3(C3_ / 128, M / 64), 128, 0, stream>>>(
        Xbf, Wqt, b_qkv, qkv, M, C3_, C_);

    // 2) V transpose per head
    {
        int n = B_ * H_ * DH_ * T_;
        k_vt<<<(n + 255) / 256, 256, 0, stream>>>(qkv, Vt);
    }

    // 3) attention
    k_attention<<<B_ * H_ * (T_ / 64), 128, 0, stream>>>(qkv, Vt, AttO);

    // 4) out = AttO * W_proj + b_proj  (fp32 out)
    k_gemm_bias<false><<<dim3(C_ / 128, M / 64), 128, 0, stream>>>(
        AttO, Wpt, b_proj, out, M, C_, C_);
}
